// MultiHeadAttention_47785806135874
// MI455X (gfx1250) — matrix operand order repaired and checked by reading, not compiled, not run on hardware
//
#include <hip/hip_runtime.h>
#include <hip/hip_bf16.h>

// Problem constants
#define BB  2
#define SS  2048
#define DDM 1024
#define HH  16
#define DKV 64

typedef _Float16 half_t;
typedef __attribute__((ext_vector_type(16))) _Float16 v16h;
typedef __attribute__((ext_vector_type(8)))  _Float16 v8h;
typedef __attribute__((ext_vector_type(8)))  float    v8f;

// ---------------------------------------------------------------------------
// WMMA helpers (CDNA5 wave32, 16x16x32 f16 -> f32 accumulate)
// A-matrix 16x32 layout: lane<16: row=lane, K={0..7,16..23}; lane>=16: row=lane-16, K={8..15,24..31}
// B-matrix 32x16 layout: lane<16: col=lane, K={0..7,16..23}; lane>=16: col=lane-16, K={8..15,24..31}
// C/D f32 16x16: VGPR r, lane<16: (M=r, N=lane); lane>=16: (M=r+8, N=lane-16)
// ---------------------------------------------------------------------------
__device__ __forceinline__ v8f wmma16(v16h a, v16h b, v8f c) {
  return __builtin_amdgcn_wmma_f32_16x16x32_f16(false, a, false, b, (short)0, c, false, false);
}

// A fragment from row-major [M][ldm] f16 matrix
__device__ __forceinline__ v16h ld_afrag(const half_t* __restrict__ base, int row,
                                         int ldm, int kofs, int lane) {
  int lr = lane & 15, hf = lane >> 4;
  const half_t* p = base + (size_t)(row + lr) * ldm + kofs + hf * 8;
  v8h lo = *(const v8h*)p;
  v8h hi = *(const v8h*)(p + 16);
  return __builtin_shufflevector(lo, hi, 0,1,2,3,4,5,6,7,8,9,10,11,12,13,14,15);
}

// B fragment from B^T stored row-major as [N][ldm] (so each lane reads two runs of 8 contiguous f16, at k and k+16)
__device__ __forceinline__ v16h ld_bfrag(const half_t* __restrict__ baseT, int col,
                                         int ldm, int kofs, int lane) {
  int lr = lane & 15, hf = lane >> 4;
  const half_t* p = baseT + (size_t)(col + lr) * ldm + kofs + hf * 8;
  v8h lo = *(const v8h*)p;
  v8h hi = *(const v8h*)(p + 16);
  return __builtin_shufflevector(lo, hi, 0,1,2,3,4,5,6,7,8,9,10,11,12,13,14,15);
}

// ---------------------------------------------------------------------------
// Kernel 0a: f32 -> f16 convert (elementwise)
// ---------------------------------------------------------------------------
__global__ void cvt_f32_f16_kernel(const float* __restrict__ src,
                                   half_t* __restrict__ dst, int n) {
  int i = blockIdx.x * blockDim.x + threadIdx.x;
  if (i < n) dst[i] = (half_t)src[i];
}

// ---------------------------------------------------------------------------
// Kernel 0b: weight transpose+convert: w[H][D][64] (f32) -> wt[(h*64+e)][D] (f16)
// ---------------------------------------------------------------------------
__global__ void wtr_kernel(const float* __restrict__ w, half_t* __restrict__ wt, int n) {
  int i = blockIdx.x * blockDim.x + threadIdx.x;
  if (i < n) {
    int h = i / (DDM * DKV);
    int rem = i - h * (DDM * DKV);
    int d = rem / DKV;
    int e = rem - d * DKV;
    wt[(size_t)(h * DKV + e) * DDM + d] = (half_t)w[i];
  }
}

// ---------------------------------------------------------------------------
// Kernel 1: fused QKV projection GEMM. M=B*S=4096 tokens, K=D=1024, N=1024 per matrix.
// block = 4 waves; wave computes a 16x64 strip. grid = (64, 48) (y: mat*16 + ntile)
// Q,K stored [b][h][s][64]; V stored transposed [b][h][64][s] for the PV WMMA.
// ---------------------------------------------------------------------------
__global__ __launch_bounds__(128) void qkv_kernel(
    const half_t* __restrict__ x,   // [B*S][D] f16
    const half_t* __restrict__ wqt, const half_t* __restrict__ wkt,
    const half_t* __restrict__ wvt, // each [H*64][D] f16 (transposed)
    half_t* __restrict__ qb, half_t* __restrict__ kb, half_t* __restrict__ vtb) {
  int lane = threadIdx.x & 31, wave = threadIdx.x >> 5;
  int lr = lane & 15, hf = lane >> 4;
  int mbase = blockIdx.x * 64 + wave * 16;
  int mat   = blockIdx.y >> 4;
  int ntile = (blockIdx.y & 15) * 64;
  const half_t* wt = (mat == 0) ? wqt : (mat == 1) ? wkt : wvt;

  v8f acc[4] = {v8f{}, v8f{}, v8f{}, v8f{}};
  for (int kd = 0; kd < DDM; kd += 32) {
    v16h a = ld_afrag(x, mbase, DDM, kd, lane);
#pragma unroll
    for (int t = 0; t < 4; ++t) {
      v16h bf = ld_bfrag(wt, ntile + t * 16, DDM, kd, lane);
      acc[t] = wmma16(a, bf, acc[t]);
    }
  }
#pragma unroll
  for (int t = 0; t < 4; ++t) {
    int n = ntile + t * 16 + lr;
    int h = n >> 6, e = n & 63;
#pragma unroll
    for (int r = 0; r < 8; ++r) {
      int tk = mbase + r + 8 * hf;
      int b = tk / SS, s = tk - b * SS;
      half_t v = (half_t)acc[t][r];
      if (mat == 0)      qb[(size_t)((b * HH + h) * SS + s) * DKV + e] = v;
      else if (mat == 1) kb[(size_t)((b * HH + h) * SS + s) * DKV + e] = v;
      else               vtb[((size_t)(b * HH + h) * DKV + e) * SS + s] = v;
    }
  }
}

// ---------------------------------------------------------------------------
// Kernel 2: attention. grid = (S/64, H*B), block = 4 waves; wave owns 16 query rows.
// Two-pass masked softmax (masked keys contribute exp=0, matching -inf masking);
// writes attns (f32) and combined context (f16, head-major concat).
// ---------------------------------------------------------------------------
__global__ __launch_bounds__(128) void attn_kernel(
    const half_t* __restrict__ qb, const half_t* __restrict__ kb,
    const half_t* __restrict__ vtb, const unsigned char* __restrict__ smask,
    float* __restrict__ attnO, half_t* __restrict__ comb) {
  __shared__ float ldsP[4 * 16 * 32];  // per-wave 16x32 probability stage
  int lane = threadIdx.x & 31, wave = threadIdx.x >> 5;
  int lr = lane & 15, hf = lane >> 4;
  int hb = blockIdx.y;                 // h*B + b (matches attns reshape)
  int h = hb / BB, b = hb - h * BB;
  int qbase = blockIdx.x * 64 + wave * 16;

  const half_t* Q  = qb  + (size_t)(b * HH + h) * SS * DKV;
  const half_t* K  = kb  + (size_t)(b * HH + h) * SS * DKV;
  const half_t* VT = vtb + (size_t)(b * HH + h) * DKV * SS;
  const unsigned char* mk = smask + b * SS;
  const float invt = 0.03125f;         // 1/sqrt(D) = 1/32

  // Q fragments kept in registers for both passes (K-dim 64 = 2 frags)
  v16h aq0 = ld_afrag(Q, qbase, DKV, 0, lane);
  v16h aq1 = ld_afrag(Q, qbase, DKV, 32, lane);

  // ---- pass 1: row sums of exp(score) (per-lane partials, one reduction) ----
  float rsum[8] = {0.f, 0.f, 0.f, 0.f, 0.f, 0.f, 0.f, 0.f};
  for (int kt = 0; kt < SS / 16; ++kt) {
    v8f c = v8f{};
    c = wmma16(aq0, ld_bfrag(K, kt * 16, DKV, 0, lane), c);
    c = wmma16(aq1, ld_bfrag(K, kt * 16, DKV, 32, lane), c);
    float mv = mk[kt * 16 + lr] ? 1.f : 0.f;
#pragma unroll
    for (int r = 0; r < 8; ++r) rsum[r] += mv * __expf(c[r] * invt);
    if (kt + 16 < SS / 16)
      __builtin_prefetch((const void*)(K + (size_t)(kt * 16 + 256) * DKV), 0, 1);
  }
  float rinv[8];
#pragma unroll
  for (int r = 0; r < 8; ++r) {
    float s = rsum[r];
#pragma unroll
    for (int i = 1; i < 16; i <<= 1) s += __shfl_xor(s, i, 32);  // half-wave reduce
    rinv[r] = (s > 0.f) ? (1.f / s) : 0.f;
  }

  // ---- pass 2: recompute scores, normalize, emit attns, accumulate P·V ----
  v8f o[4] = {v8f{}, v8f{}, v8f{}, v8f{}};
  float* myP = ldsP + wave * (16 * 32);
  for (int kc = 0; kc < SS / 32; ++kc) {
#pragma unroll
    for (int sub = 0; sub < 2; ++sub) {
      int kn = kc * 32 + sub * 16;
      v8f c = v8f{};
      c = wmma16(aq0, ld_bfrag(K, kn, DKV, 0, lane), c);
      c = wmma16(aq1, ld_bfrag(K, kn, DKV, 32, lane), c);
      float mv = mk[kn + lr] ? 1.f : 0.f;
#pragma unroll
      for (int r = 0; r < 8; ++r) {
        float p = mv * __expf(c[r] * invt) * rinv[r];
        int q = qbase + r + 8 * hf;
        attnO[((size_t)hb * SS + q) * SS + kn + lr] = p;
        myP[(r + 8 * hf) * 32 + sub * 16 + lr] = p;
      }
    }
    __syncthreads();
    // rebuild P as an A-fragment (16 queries x 32 keys) from LDS
    v16h ap;
#pragma unroll
    for (int j = 0; j < 8; ++j) {
      ap[j]     = (half_t)myP[lr * 32 + hf * 8 + j];
      ap[j + 8] = (half_t)myP[lr * 32 + hf * 8 + 16 + j];
    }
#pragma unroll
    for (int t = 0; t < 4; ++t)
      o[t] = wmma16(ap, ld_bfrag(VT, t * 16, SS, kc * 32, lane), o[t]);
    __syncthreads();
  }

  // combined[b][s][h*64 + dv] (head-major concat), f16 for the final GEMM
#pragma unroll
  for (int t = 0; t < 4; ++t) {
    int dv = t * 16 + lr;
#pragma unroll
    for (int r = 0; r < 8; ++r) {
      int q = qbase + r + 8 * hf;
      comb[((size_t)b * SS + q) * (HH * DKV) + h * DKV + dv] = (half_t)o[t][r];
    }
  }
}

// ---------------------------------------------------------------------------
// Kernel 3: output projection + residual. out[t][d] = sum_c comb[t][c]*proj_w[d][c]
// proj_w natural layout [d][c] is exactly B^T -> contiguous B fragments.
// ---------------------------------------------------------------------------
__global__ __launch_bounds__(128) void proj_kernel(
    const half_t* __restrict__ comb, const half_t* __restrict__ pw,
    const float* __restrict__ inp, float* __restrict__ y) {
  int lane = threadIdx.x & 31, wave = threadIdx.x >> 5;
  int lr = lane & 15, hf = lane >> 4;
  int mbase = blockIdx.x * 64 + wave * 16;
  int ntile = blockIdx.y * 64;

  v8f acc[4] = {v8f{}, v8f{}, v8f{}, v8f{}};
  for (int kd = 0; kd < DDM; kd += 32) {
    v16h a = ld_afrag(comb, mbase, DDM, kd, lane);
#pragma unroll
    for (int t = 0; t < 4; ++t)
      acc[t] = wmma16(a, ld_bfrag(pw, ntile + t * 16, DDM, kd, lane), acc[t]);
  }
#pragma unroll
  for (int t = 0; t < 4; ++t) {
    int n = ntile + t * 16 + lr;
#pragma unroll
    for (int r = 0; r < 8; ++r) {
      int tk = mbase + r + 8 * hf;
      y[(size_t)tk * DDM + n] = acc[t][r] + inp[(size_t)tk * DDM + n];
    }
  }
}

// ---------------------------------------------------------------------------
// Kernel 4: LayerNorm with unbiased std (torch ddof=1), eps added to sigma.
// One block (256 threads = 8 waves) per token row.
// ---------------------------------------------------------------------------
__global__ __launch_bounds__(256) void ln_kernel(
    const float* __restrict__ y, const float* __restrict__ ga,
    const float* __restrict__ be, float* __restrict__ out) {
  __shared__ float red[2][8];
  int t = blockIdx.x;
  const float* row = y + (size_t)t * DDM;
  float s = 0.f, s2 = 0.f;
  for (int i = threadIdx.x; i < DDM; i += 256) {
    float v = row[i];
    s += v; s2 += v * v;
  }
#pragma unroll
  for (int i = 1; i < 32; i <<= 1) { s += __shfl_xor(s, i, 32); s2 += __shfl_xor(s2, i, 32); }
  int wv = threadIdx.x >> 5, ln = threadIdx.x & 31;
  if (ln == 0) { red[0][wv] = s; red[1][wv] = s2; }
  __syncthreads();
  if (wv == 0) {
    s  = (ln < 8) ? red[0][ln] : 0.f;
    s2 = (ln < 8) ? red[1][ln] : 0.f;
#pragma unroll
    for (int i = 1; i < 8; i <<= 1) { s += __shfl_xor(s, i, 32); s2 += __shfl_xor(s2, i, 32); }
    if (ln == 0) { red[0][0] = s; red[1][0] = s2; }
  }
  __syncthreads();
  s = red[0][0]; s2 = red[1][0];
  float mu  = s / (float)DDM;
  float var = (s2 - (float)DDM * mu * mu) / (float)(DDM - 1);
  float sig = sqrtf(fmaxf(var, 0.f));
  float inv = 1.f / (sig + 1e-3f);
  for (int i = threadIdx.x; i < DDM; i += 256)
    out[(size_t)t * DDM + i] = (row[i] - mu) * inv * ga[i] + be[i];
}

// ---------------------------------------------------------------------------
// Launch
// ---------------------------------------------------------------------------
extern "C" void kernel_launch(void* const* d_in, const int* in_sizes, int n_in,
                              void* d_out, int out_size, void* d_ws, size_t ws_size,
                              hipStream_t stream) {
  (void)in_sizes; (void)n_in; (void)out_size; (void)ws_size;
  const float* inp            = (const float*)d_in[0];
  const unsigned char* smask  = (const unsigned char*)d_in[1];  // bool mask [B,S]
  const float* w_qs           = (const float*)d_in[2];
  const float* w_ks           = (const float*)d_in[3];
  const float* w_vs           = (const float*)d_in[4];
  const float* proj_w         = (const float*)d_in[5];
  const float* ln_a           = (const float*)d_in[6];
  const float* ln_b           = (const float*)d_in[7];

  float* out_ln   = (float*)d_out;                         // [B,S,D]
  float* out_attn = out_ln + (size_t)BB * SS * DDM;        // [H*B,S,S]

  // workspace carve-up
  char* ws = (char*)d_ws;
  size_t off = 0;
  auto carve = [&](size_t bytes) -> void* {
    void* p = ws + off;
    off += (bytes + 255) & ~(size_t)255;
    return p;
  };
  half_t* x_h  = (half_t*)carve((size_t)BB * SS * DDM * sizeof(half_t));
  half_t* wqt  = (half_t*)carve((size_t)HH * DDM * DKV * sizeof(half_t));
  half_t* wkt  = (half_t*)carve((size_t)HH * DDM * DKV * sizeof(half_t));
  half_t* wvt  = (half_t*)carve((size_t)HH * DDM * DKV * sizeof(half_t));
  half_t* pwh  = (half_t*)carve((size_t)DDM * HH * DKV * sizeof(half_t));
  half_t* qb   = (half_t*)carve((size_t)BB * HH * SS * DKV * sizeof(half_t));
  half_t* kbuf = (half_t*)carve((size_t)BB * HH * SS * DKV * sizeof(half_t));
  half_t* vtb  = (half_t*)carve((size_t)BB * HH * DKV * SS * sizeof(half_t));
  half_t* comb = (half_t*)carve((size_t)BB * SS * HH * DKV * sizeof(half_t));
  float*  ybuf = (float*)carve((size_t)BB * SS * DDM * sizeof(float));

  // 0) precision conversion / weight transposes
  int nx = BB * SS * DDM;
  cvt_f32_f16_kernel<<<(nx + 255) / 256, 256, 0, stream>>>(inp, x_h, nx);
  int nw = HH * DDM * DKV;
  wtr_kernel<<<(nw + 255) / 256, 256, 0, stream>>>(w_qs, wqt, nw);
  wtr_kernel<<<(nw + 255) / 256, 256, 0, stream>>>(w_ks, wkt, nw);
  wtr_kernel<<<(nw + 255) / 256, 256, 0, stream>>>(w_vs, wvt, nw);
  int np = DDM * HH * DKV;
  cvt_f32_f16_kernel<<<(np + 255) / 256, 256, 0, stream>>>(proj_w, pwh, np);

  // 1) QKV projections (WMMA)
  qkv_kernel<<<dim3((BB * SS) / 64, 48), 128, 0, stream>>>(x_h, wqt, wkt, wvt,
                                                           qb, kbuf, vtb);
  // 2) attention (WMMA scores + WMMA P·V), writes attns + combined
  attn_kernel<<<dim3(SS / 64, HH * BB), 128, 0, stream>>>(qb, kbuf, vtb, smask,
                                                          out_attn, comb);
  // 3) output projection + residual (WMMA)
  proj_kernel<<<dim3((BB * SS) / 64, DDM / 64), 128, 0, stream>>>(comb, pwh, inp, ybuf);
  // 4) layernorm
  ln_kernel<<<BB * SS, 256, 0, stream>>>(ybuf, ln_a, ln_b, out_ln);
}
